// PredCell_47991964565621
// MI455X (gfx1250) — matrix operand order repaired and checked by reading, not compiled, not run on hardware
//
#include <hip/hip_runtime.h>
#include <hip/hip_bf16.h>

// ---------------------------------------------------------------------------
// PredCell persistent kernel for MI455X (gfx1250, wave32, WMMA).
//
// Batch-1 matvec chain, weight-bandwidth bound:
//  * weights packed to bf16 in d_ws (93 MB -> resident in 192 MB L2)
//  * one persistent kernel, in-kernel grid barrier (atomic counter + s_sleep)
//  * matvecs via v_wmma_f32_16x16x32_bf16; B operand = x broadcast to all
//    16 columns (broadcast ds reads from LDS); software-pipelined K loop
//    (double-buffered fragments) so A-loads overlap WMMA issue
//  * x vectors staged into LDS with GLOBAL_LOAD_ASYNC_TO_LDS_B128 (ASYNCcnt)
//    when the toolchain exposes the builtin; plain copy fallback otherwise
//  * LSTM f-gate skipped (c0 == 0 makes it dead in the reference)
//  * all epilogue decisions are template parameters -> straight-line code
// Requires ws_size >= ~92.6 MB and 32 co-resident workgroups.
// ---------------------------------------------------------------------------

#define T_STEPS 512
#define NCHARS  56
#define NBLK    32
#define TPB     256
#define NWAVES  (NBLK * (TPB / 32))   // 256 waves

typedef __bf16 bf16;
typedef __attribute__((ext_vector_type(16))) __bf16 v16bf;
typedef __attribute__((ext_vector_type(8)))  float  v8f;
typedef __attribute__((ext_vector_type(4)))  int    v4i;

#define AS(n) __attribute__((address_space(n)))

union FragB { v16bf v; uint4 q[2]; };

// ---- float workspace region (element offsets) ----
constexpr int F_REC0  = 0;                    // 64 (56 valid)  rec[0]
constexpr int F_REC14 = 64;                   // 4*1024         rec[1..4]
constexpr int F_G     = F_REC14 + 4 * 1024;   // 4096           LSTM preacts
constexpr int F_LOSS  = F_G + 4096;
constexpr int F_CNT   = F_LOSS + 1;           // grid-barrier counter (u32)
constexpr int F_TOTAL = 8320;                 // padded, 16B aligned
constexpr size_t F_BYTES = (size_t)F_TOTAL * 4;

// ---- bf16 workspace region (element offsets from wsb) ----
constexpr size_t B_TD0B   = 0;                        // 64 (padded from 56)
constexpr size_t B_XIN1   = 64;                       // 2048: [bu0 | td[0]]
constexpr size_t B_XIN2   = B_XIN1 + 2048;
constexpr size_t B_XIN3   = B_XIN2 + 2048;
constexpr size_t B_XIN4   = B_XIN3 + 2048;
constexpr size_t B_XIN5   = B_XIN4 + 2048;            // 1024: bu only (top)
constexpr size_t B_STATEB = B_XIN5 + 1024;            // 1024
constexpr size_t B_ACTEND = B_STATEB + 1024;          // 10304
constexpr size_t B_WW0    = B_ACTEND;                 // 1024 x 64 (K padded)
constexpr size_t B_WW14   = B_WW0   + 1024 * 64;      // 4 x 1024x1024
constexpr size_t B_WIH03  = B_WW14  + 4ull * 1024 * 1024;   // 4 x 4096x2048
constexpr size_t B_WIH4   = B_WIH03 + 4ull * 4096 * 2048;   // 4096x1024
constexpr size_t B_VW0    = B_WIH4  + 4096ull * 1024; // 64 x 1024 (R padded)
constexpr size_t B_VW14   = B_VW0   + 64 * 1024;      // 4 x 1024x1024
constexpr size_t B_END    = B_VW14  + 4ull * 1024 * 1024;   // 46,278,720 elems

struct KParams {
  const float* X;          // [512, 56]
  const int*   iter;       // scalar
  const float* Vb[5];
  const float* Vw[5];
  const float* Wb[5];
  const float* Ww[5];
  const float* bhh[5];
  const float* bih[5];
  const float* wih[5];
  float* wsf;              // fp32 ws region
  bf16*  wsb;              // bf16 ws region
  float* out;              // d_out (1 float)
};

// ---------------------------------------------------------------------------
// Grid-wide barrier: monotonic counter, release/acquire fences, s_sleep spin.
// All blocks call this the same number of times (uniform control flow).
// ---------------------------------------------------------------------------
__device__ __forceinline__ void gsync(unsigned* cnt, unsigned& bar) {
  __threadfence();            // release: flush our global stores to L2
  __syncthreads();
  ++bar;
  if (threadIdx.x == 0) {
    __hip_atomic_fetch_add(cnt, 1u, __ATOMIC_RELEASE, __HIP_MEMORY_SCOPE_AGENT);
    while (__hip_atomic_load(cnt, __ATOMIC_ACQUIRE, __HIP_MEMORY_SCOPE_AGENT)
           < (unsigned)NBLK * bar) {
      __builtin_amdgcn_s_sleep(8);
    }
  }
  __syncthreads();
  __threadfence();            // acquire: invalidate L0 so we see other WGPs
}

// fp32 -> bf16 pack with row/col zero padding (row-major R x K -> Rp x Kp)
__device__ __forceinline__ void pack_mat(const float* __restrict__ src,
                                         bf16* __restrict__ dst,
                                         int R, int K, int Kp, int Rp,
                                         int gtid, int gsz) {
  const int n = Rp * Kp;
  for (int i = gtid; i < n; i += gsz) {
    const int r = i / Kp, k = i - r * Kp;
    float v = (r < R && k < K) ? src[r * K + k] : 0.0f;
    dst[i] = (bf16)v;
  }
}

// Stage n bf16 (n % 8 == 0, 16B-aligned) from global into LDS.
// Preferred path: GLOBAL_LOAD_ASYNC_TO_LDS_B128 (CDNA5 async DMA, ASYNCcnt);
// each wave drains its own ASYNCcnt before the block barrier that follows.
__device__ __forceinline__ void stage_lds(bf16* __restrict__ s,
                                          const bf16* __restrict__ g,
                                          int n, int tid) {
#if __has_builtin(__builtin_amdgcn_global_load_async_to_lds_b128)
  const int chunks = n >> 3;                      // 16 bytes per chunk
  for (int i = tid; i < chunks; i += TPB) {
    __builtin_amdgcn_global_load_async_to_lds_b128(
        (AS(1) v4i*)(g + (size_t)i * 8), (AS(3) v4i*)(s + (size_t)i * 8),
        0, 0);
  }
  asm volatile("s_wait_asynccnt 0" ::: "memory");
#else
  for (int i = tid; i < n; i += TPB) s[i] = g[i];
#endif
}

// fragment loads
// A (16-bit A 16x32): lane half -> K base+0..15 contiguous (this code's order, not ISA table)
__device__ __forceinline__ void ld_a(FragB& f, const bf16* p) {
  f.q[0] = *(const uint4*)(p);
  f.q[1] = *(const uint4*)(p + 8);
}
// B (16-bit B 32x16): lane half -> K base+0..15 contiguous (this code's order, not ISA table)
__device__ __forceinline__ void ld_b(FragB& f, const bf16* p) {
  f.q[0] = *(const uint4*)(p);
  f.q[1] = *(const uint4*)(p + 8);
}

__device__ __forceinline__ v8f wmma_bf16(const FragB& a, const FragB& b, v8f c) {
  return __builtin_amdgcn_wmma_f32_16x16x32_bf16(false, a.v, false, b.v,
                                                 (short)0, c, false, false);
}

// ---------------------------------------------------------------------------
// WMMA matvec: y[0:rows_valid] = W(Rp x K, bf16, row-major) @ x + b0 (+ b1)
// One 16-row tile per wave iteration; every D column equals the y-tile, so
// lanes (lane&15)==0 write the 16 results. K-loop is software-pipelined with
// double-buffered fragments (no conditionals in the steady-state body).
// ---------------------------------------------------------------------------
template <int K, bool HASB1, bool OUTF, bool OUTB, bool SKIPF, bool CHECKR>
__device__ __forceinline__ void mv_wmma(const bf16* __restrict__ W,
                                        const bf16* __restrict__ sx,  // LDS
                                        const float* __restrict__ b0,
                                        const float* __restrict__ b1,
                                        float* __restrict__ of,
                                        bf16* __restrict__ ob,
                                        int tile0, int ntiles, int tstride,
                                        int rows_valid, int lane) {
  static_assert(K % 64 == 0, "K must be a multiple of 64");
  const int hi  = lane >> 4;        // lane half
  const int rm  = lane & 15;        // row within tile (A), col (B/D)
  const int klA = hi * 16;          // A-frag K base
  const int klB = hi * 16;          // B-frag K base
  for (int t = tile0; t < ntiles; t += tstride) {
    const int rt = SKIPF ? (t < 64 ? t : t + 64) : t;   // skip LSTM f-gate
    const bf16* wrow = W + (size_t)(rt * 16 + rm) * (size_t)K + klA;
    const bf16* xv   = sx + klB;
    FragB a0, b0f, a1, b1f;
    ld_a(a0, wrow);
    ld_b(b0f, xv);
    v8f acc = {};
    for (int k0 = 0; k0 < K - 64; k0 += 64) {
      ld_a(a1, wrow + k0 + 32);
      ld_b(b1f, xv + k0 + 32);
      acc = wmma_bf16(a0, b0f, acc);
      ld_a(a0, wrow + k0 + 64);
      ld_b(b0f, xv + k0 + 64);
      acc = wmma_bf16(a1, b1f, acc);
    }
    ld_a(a1, wrow + (K - 32));
    ld_b(b1f, xv + (K - 32));
    acc = wmma_bf16(a0, b0f, acc);
    acc = wmma_bf16(a1, b1f, acc);

    if (rm == 0) {
      const int rb = rt * 16 + hi * 8;
#pragma unroll
      for (int v = 0; v < 8; ++v) {
        const int r = rb + v;
        if (CHECKR && r >= rows_valid) continue;
        float y = acc[v] + b0[r];
        if (HASB1) y += b1[r];
        if (OUTF) of[r] = y;
        if (OUTB) ob[r] = (bf16)y;
      }
    }
  }
}

__device__ __forceinline__ float sigm(float x) {
  return 1.0f / (1.0f + __expf(-x));
}

__device__ __forceinline__ void wave_loss(float a, float* lossp, int lane) {
  for (int o = 16; o; o >>= 1) a += __shfl_down(a, o, 32);
  if (lane == 0 && a != 0.0f) atomicAdd(lossp, a);
}

// layer-0 ErrorUnit: td0 = x_t - rec0 (56 valid, padded to 64), loss += |td0|
__device__ __forceinline__ void e0_stage(const float* __restrict__ xt,
                                         const float* __restrict__ rec0,
                                         bf16* __restrict__ td0b,
                                         float* lossp, int i, int lane) {
  const float xv = (i < NCHARS) ? xt[i]   : 0.0f;
  const float rv = (i < NCHARS) ? rec0[i] : 0.0f;
  const float td = xv - rv;
  td0b[i] = (bf16)td;
  wave_loss(fabsf(td), lossp, lane);
}

// fused: state = sig(o)*tanh(sig(i)*tanh(c)); if TD: td = state - rec_next
template <bool TD>
__device__ __forceinline__ void act_stage(const float* __restrict__ g,
                                          const float* __restrict__ rec_next,
                                          bf16* __restrict__ stateb,
                                          bf16* __restrict__ tdb,
                                          float* lossp, float lampow,
                                          int i, int lane) {
  float a = 0.0f;
  if (i < 1024) {
    const float gi = g[i];            // i-gate
    const float gc = g[i + 2048];     // c(g)-gate  (f-gate region skipped)
    const float go = g[i + 3072];     // o-gate
    const float cell = sigm(gi) * tanhf(gc);
    const float st   = sigm(go) * tanhf(cell);
    stateb[i] = (bf16)st;
    if (TD) {
      const float td = st - rec_next[i];
      tdb[i] = (bf16)td;
      a = fabsf(td) * lampow;
    }
  }
  if (TD) wave_loss(a, lossp, lane);
}

// ---------------------------------------------------------------------------
__global__ void init_kernel(KParams p) {
  const int i = blockIdx.x * blockDim.x + threadIdx.x;
  const int n = gridDim.x * blockDim.x;
  for (int k = i; k < F_TOTAL; k += n) p.wsf[k] = 0.0f;
  for (size_t k = i; k < B_ACTEND; k += n) p.wsb[k] = (bf16)0.0f;
}

__global__ void __launch_bounds__(TPB, 1) predcell_kernel(KParams p) {
  __shared__ __align__(16) bf16 sx[4096];

  const int tid  = threadIdx.x;
  const int bid  = blockIdx.x;
  const int lane = tid & 31;
  const int gw   = bid * (TPB / 32) + (tid >> 5);   // global wave id
  const int gtid = bid * TPB + tid;
  const int gsz  = NBLK * TPB;
  unsigned bar = 0;

  float* wsf   = p.wsf;
  bf16*  wsb   = p.wsb;
  float* lossp = wsf + F_LOSS;
  unsigned* cnt = (unsigned*)(wsf + F_CNT);

  float* recf[5];
  recf[0] = wsf + F_REC0;
  for (int k = 1; k < 5; ++k) recf[k] = wsf + F_REC14 + (k - 1) * 1024;
  float* gbuf = wsf + F_G;

  bf16* xin[5] = { wsb + B_XIN1, wsb + B_XIN2, wsb + B_XIN3,
                   wsb + B_XIN4, wsb + B_XIN5 };
  bf16* stateb = wsb + B_STATEB;
  bf16* td0b   = wsb + B_TD0B;

  bf16* wbWw[5];
  wbWw[0] = wsb + B_WW0;
  for (int k = 1; k < 5; ++k) wbWw[k] = wsb + B_WW14 + (size_t)(k - 1) * 1024 * 1024;
  bf16* wbWih[5];
  for (int k = 0; k < 4; ++k) wbWih[k] = wsb + B_WIH03 + (size_t)k * 4096 * 2048;
  wbWih[4] = wsb + B_WIH4;
  bf16* wbVw[5];
  wbVw[0] = wsb + B_VW0;
  for (int k = 1; k < 5; ++k) wbVw[k] = wsb + B_VW14 + (size_t)(k - 1) * 1024 * 1024;

  // -------- prologue: pack weights fp32 -> bf16 (L2-resident model) --------
  pack_mat(p.Ww[0], wbWw[0], 1024, 56, 64, 1024, gtid, gsz);
  for (int m = 1; m < 5; ++m)
    pack_mat(p.Ww[m], wbWw[m], 1024, 1024, 1024, 1024, gtid, gsz);
  for (int m = 0; m < 4; ++m)
    pack_mat(p.wih[m], wbWih[m], 4096, 2048, 2048, 4096, gtid, gsz);
  pack_mat(p.wih[4], wbWih[4], 4096, 1024, 1024, 4096, gtid, gsz);
  pack_mat(p.Vw[0], wbVw[0], 56, 1024, 1024, 64, gtid, gsz);
  for (int m = 1; m < 5; ++m)
    pack_mat(p.Vw[m], wbVw[m], 1024, 1024, 1024, 1024, gtid, gsz);
  // layer-0 error for t=0 (rec0 zeroed by init kernel)
  if (bid == 0 && tid < 64) e0_stage(p.X, recf[0], td0b, lossp, tid, lane);
  gsync(cnt, bar);

  const int it = *p.iter;
  const float lam = (it <= 1000) ? 1e-4f : 1e-2f;
  float lp[5];
  lp[0] = 1.0f;
  for (int l = 1; l < 5; ++l) lp[l] = lp[l - 1] * lam;

  for (int t = 0; t < T_STEPS; ++t) {
    // ---- stage: bu0 = Ww0 @ td0 + Wb0  -> xin1[0:1024] (bf16) ----
    stage_lds(sx, td0b, 64, tid);
    __syncthreads();
    if (gw < 64)
      mv_wmma<64, false, false, true, false, false>(
          wbWw[0], sx, p.Wb[0], nullptr, nullptr, xin[0], gw, 64, 64, 1024, lane);
    gsync(cnt, bar);

    // ================= layers 1..4 (K = 2048, TD + bu path) =================
    for (int lyr = 1; lyr <= 4; ++lyr) {
      const int j = lyr - 1;

      // ---- stage: g = wih[j] @ xin + bih + bhh (f-gate skipped) ----
      stage_lds(sx, xin[j], 2048, tid);
      __syncthreads();
      mv_wmma<2048, true, true, false, true, false>(
          wbWih[j], sx, p.bih[j], p.bhh[j], gbuf, nullptr,
          gw, 192, NWAVES, 4096, lane);
      gsync(cnt, bar);

      // ---- stage: activation + td_l + loss ----
      act_stage<true>(gbuf, recf[lyr], stateb, xin[j] + 1024, lossp,
                      lp[lyr], gtid, lane);
      gsync(cnt, bar);

      // ---- stage: rec[j] = Vw@state  ||  bu = Ww[lyr]@td ----
      stage_lds(sx, stateb, 1024, tid);
      stage_lds(sx + 1024, xin[j] + 1024, 1024, tid);
      __syncthreads();
      if (gw < 64) {
        if (j == 0)
          mv_wmma<1024, false, true, false, false, true>(
              wbVw[0], sx, p.Vb[0], nullptr, recf[0], nullptr,
              gw, 4, 64, NCHARS, lane);
        else
          mv_wmma<1024, false, true, false, false, false>(
              wbVw[j], sx, p.Vb[j], nullptr, recf[j], nullptr,
              gw, 64, 64, 1024, lane);
      } else if (gw < 128) {
        mv_wmma<1024, false, false, true, false, false>(
            wbWw[lyr], sx + 1024, p.Wb[lyr], nullptr, nullptr, xin[lyr],
            gw - 64, 64, 64, 1024, lane);
      }
      gsync(cnt, bar);
    }

    // ======================= layer 5 (K = 1024, top) ========================
    stage_lds(sx, xin[4], 1024, tid);
    __syncthreads();
    mv_wmma<1024, true, true, false, true, false>(
        wbWih[4], sx, p.bih[4], p.bhh[4], gbuf, nullptr,
        gw, 192, NWAVES, 4096, lane);
    gsync(cnt, bar);

    act_stage<false>(gbuf, nullptr, stateb, nullptr, lossp, 0.0f, gtid, lane);
    gsync(cnt, bar);

    stage_lds(sx, stateb, 1024, tid);
    __syncthreads();
    if (gw < 64) {
      mv_wmma<1024, false, true, false, false, false>(
          wbVw[4], sx, p.Vb[4], nullptr, recf[4], nullptr,
          gw, 64, 64, 1024, lane);
    } else if ((t + 1) < T_STEPS && bid == 8 && tid < 64) {
      // fused layer-0 error for the NEXT timestep (rec0 is final for step t)
      e0_stage(p.X + (size_t)(t + 1) * NCHARS, recf[0], td0b, lossp, tid, lane);
    }
    gsync(cnt, bar);
  }

  if (gtid == 0) p.out[0] = wsf[F_LOSS];
}

// ---------------------------------------------------------------------------
extern "C" void kernel_launch(void* const* d_in, const int* in_sizes, int n_in,
                              void* d_out, int out_size, void* d_ws, size_t ws_size,
                              hipStream_t stream) {
  (void)in_sizes; (void)n_in; (void)out_size; (void)ws_size;

  KParams p;
  p.X    = (const float*)d_in[0];      // input_sentence [512,56]
  p.iter = (const int*)d_in[1];        // iternumber
  // params dict flattened as a JAX pytree: dict keys in sorted order:
  //   Vb, Vw, Wb, Ww, bhh, bih, wih   (each a list, in index order)
  int k = 2;
  for (int i = 0; i < 5; ++i) p.Vb[i]  = (const float*)d_in[k++];
  for (int i = 0; i < 5; ++i) p.Vw[i]  = (const float*)d_in[k++];
  for (int i = 0; i < 5; ++i) p.Wb[i]  = (const float*)d_in[k++];
  for (int i = 0; i < 5; ++i) p.Ww[i]  = (const float*)d_in[k++];
  for (int i = 0; i < 5; ++i) p.bhh[i] = (const float*)d_in[k++];
  for (int i = 0; i < 5; ++i) p.bih[i] = (const float*)d_in[k++];
  for (int i = 0; i < 5; ++i) p.wih[i] = (const float*)d_in[k++];

  p.wsf = (float*)d_ws;
  p.wsb = (bf16*)((char*)d_ws + F_BYTES);   // needs ~92.6 MB total workspace
  p.out = (float*)d_out;

  hipLaunchKernelGGL(init_kernel, dim3(64), dim3(256), 0, stream, p);
  hipLaunchKernelGGL(predcell_kernel, dim3(NBLK), dim3(TPB), 0, stream, p);
}
